// ScaledDotProductAttention_27109833572478
// MI455X (gfx1250) — matrix operand order repaired and checked by reading, not compiled, not run on hardware
//
#include <hip/hip_runtime.h>

// ---------------------------------------------------------------------------
// Flash-attention forward, B=2,H=16,S=2048,D=64 fp32, causal.
// v_wmma_f32_16x16x32_f16 with f32 accumulation, online softmax (exp2 domain),
// row-sum via ones-matrix WMMA, row-max via ds_swizzle xor reduction,
// double-buffered K/V LDS tiles (one barrier per key block),
// mask applied only in the two diagonal blocks, per-wave tail skip.
// ---------------------------------------------------------------------------

typedef _Float16 v16h __attribute__((ext_vector_type(16)));
typedef _Float16 v8h  __attribute__((ext_vector_type(8)));
typedef _Float16 h2   __attribute__((ext_vector_type(2)));
typedef float    v8f  __attribute__((ext_vector_type(8)));
typedef float    v4f  __attribute__((ext_vector_type(4)));

#define S_LEN  2048
#define D_DIM  64
#define NWAVE  4
#define QTILE  (NWAVE * 16)   // 64 query rows per workgroup
#define KTILE  32             // keys per inner iteration
#define KSTR   80             // halves per key row in sK   (16B-aligned rows)
#define VSTR   40             // halves per dim row in sV^T (16B-aligned rows)
#define PSTR   40             // halves per row in sP

// (1/8)*log2(e): scores land directly in log2 domain.
#define QK_SCALE 0.1803368801111204f

// ds_swizzle_b32 immediate: xor_mask in [14:10], and_mask=0x1F (group-of-32).
#define SWZ_XOR(x, m) \
  __int_as_float(__builtin_amdgcn_ds_swizzle(__float_as_int(x), (0x1F | ((m) << 10))))

__device__ __forceinline__ v8f wmma_f16(v16h a, v16h b, v8f c) {
  return __builtin_amdgcn_wmma_f32_16x16x32_f16(false, a, false, b, (short)0, c,
                                                false, false);
}

union V16 { v16h v; v8h h[2]; };

__global__ __launch_bounds__(128) void flash_attn_causal_f16wmma(
    const float* __restrict__ Q, const float* __restrict__ K,
    const float* __restrict__ V, float* __restrict__ Out) {
  __shared__ __align__(16) _Float16 sK[2][KTILE * KSTR];   // [key][dim] f16
  __shared__ __align__(16) _Float16 sV[2][D_DIM * VSTR];   // [dim][key] f16
  __shared__ __align__(16) _Float16 sP[NWAVE][16 * PSTR];  // per-wave P tile

  const int tid  = threadIdx.x;
  const int lane = tid & 31;
  const int wave = tid >> 5;
  const int col  = lane & 15;
  const int hi   = lane >> 4;

  const int    bh    = blockIdx.y;
  const int    qblk  = blockIdx.x * QTILE;
  const size_t base  = (size_t)bh * S_LEN * D_DIM;
  const int    qrow0 = qblk + wave * 16;

  // ---- Q fragments (A layout), fp32 -> f16, scale folded in ----
  V16 qa[2];
  {
    const float* qp = Q + base + (size_t)(qrow0 + col) * D_DIM;
#pragma unroll
    for (int c = 0; c < 2; ++c) {
#pragma unroll
      for (int h = 0; h < 2; ++h) {
        const int d0 = c * 32 + hi * 16 + h * 8;
        v4f f0 = *(const v4f*)(qp + d0);
        v4f f1 = *(const v4f*)(qp + d0 + 4);
        v8h hh;
#pragma unroll
        for (int i = 0; i < 4; ++i) {
          hh[i]     = (_Float16)(f0[i] * QK_SCALE);
          hh[i + 4] = (_Float16)(f1[i] * QK_SCALE);
        }
        qa[c].h[h] = hh;
      }
    }
  }

  // All-ones B fragment for row-sum WMMA.
  v16h ones;
#pragma unroll
  for (int i = 0; i < 16; ++i) ones[i] = (_Float16)1.0f;

  float m_r[8], l_r[8];
  v8f oacc[4];
#pragma unroll
  for (int r = 0; r < 8; ++r) { m_r[r] = -1e30f; l_r[r] = 0.0f; }
#pragma unroll
  for (int n = 0; n < 4; ++n)
#pragma unroll
    for (int i = 0; i < 8; ++i) oacc[n][i] = 0.0f;

  const float* kg = K + base;
  const float* vg = V + base;
  const int nkt = qblk / KTILE + 2;  // causal: keys up to qblk+63

  // Per-thread staging geometry for cooperative tile loads.
  const int kkey = tid & 31, kdseg = tid >> 5;        // K: 1 key x 16 dims
  const int vkp  = (tid & 15) * 2, vdseg = tid >> 4;  // V: 2 keys x 8 dims

  v4f kst[4], vst0[2], vst1[2];
  auto issue_loads = [&](int k0) {
    const float* ks = kg + (size_t)(k0 + kkey) * D_DIM + kdseg * 16;
#pragma unroll
    for (int i = 0; i < 4; ++i) kst[i] = *(const v4f*)(ks + i * 4);
    const float* vs = vg + (size_t)(k0 + vkp) * D_DIM + vdseg * 8;
#pragma unroll
    for (int i = 0; i < 2; ++i) {
      vst0[i] = *(const v4f*)(vs + i * 4);
      vst1[i] = *(const v4f*)(vs + D_DIM + i * 4);
    }
  };
  auto stage_tile = [&](int buf) {
    _Float16* dk = &sK[buf][kkey * KSTR + kdseg * 16];
    v8h h0, h1;
#pragma unroll
    for (int i = 0; i < 4; ++i)
#pragma unroll
      for (int j = 0; j < 4; ++j) {
        _Float16 x = (_Float16)kst[i][j];
        if (i < 2) h0[i * 4 + j] = x; else h1[(i - 2) * 4 + j] = x;
      }
    *(v8h*)dk = h0;
    *(v8h*)(dk + 8) = h1;
    _Float16* dv = &sV[buf][(vdseg * 8) * VSTR + vkp];
#pragma unroll
    for (int i = 0; i < 8; ++i) {
      h2 p;
      p[0] = (_Float16)vst0[i >> 2][i & 3];
      p[1] = (_Float16)vst1[i >> 2][i & 3];
      *(h2*)(dv + i * VSTR) = p;
    }
  };

  issue_loads(0);
  stage_tile(0);

  for (int kt = 0; kt < nkt; ++kt) {
    const int k0  = kt * KTILE;
    const int buf = kt & 1;

    if (kt + 1 < nkt) issue_loads(k0 + KTILE);  // HBM latency overlaps compute
    __syncthreads();                            // tile kt visible to all waves

    // Wave entirely above the diagonal for this block -> all P would be 0.
    if (k0 <= qrow0 + 15) {
      const _Float16* bKp = sK[buf];
      const _Float16* bVp = sV[buf];

      // ---- Preload all 4 K B-fragments, then 4 WMMAs (partial dscnt waits) ----
      V16 bk[4];  // [c*2 + kh]
#pragma unroll
      for (int c = 0; c < 2; ++c)
#pragma unroll
        for (int kh = 0; kh < 2; ++kh) {
          const _Float16* p = bKp + (kh * 16 + col) * KSTR + c * 32 + hi * 16;
          bk[c * 2 + kh].h[0] = *(const v8h*)(p);
          bk[c * 2 + kh].h[1] = *(const v8h*)(p + 8);
        }
      v8f s0a, s1a;
#pragma unroll
      for (int i = 0; i < 8; ++i) { s0a[i] = 0.0f; s1a[i] = 0.0f; }
      s0a = wmma_f16(qa[0].v, bk[0].v, s0a);
      s1a = wmma_f16(qa[0].v, bk[1].v, s1a);
      s0a = wmma_f16(qa[1].v, bk[2].v, s0a);
      s1a = wmma_f16(qa[1].v, bk[3].v, s1a);

      // ---- Mask only in the two diagonal blocks (uniform branch) ----
      float x0[8], x1[8], mx[8];
#pragma unroll
      for (int r = 0; r < 8; ++r) { x0[r] = s0a[r]; x1[r] = s1a[r]; }
      if (kt >= nkt - 2) {
        const int key_a = k0 + col, key_b = key_a + 16;
#pragma unroll
        for (int r = 0; r < 8; ++r) {
          const int rowg = qrow0 + r + 8 * hi;
          if (key_a > rowg) x0[r] = -1e30f;
          if (key_b > rowg) x1[r] = -1e30f;
        }
      }
#pragma unroll
      for (int r = 0; r < 8; ++r) mx[r] = fmaxf(x0[r], x1[r]);
#pragma unroll
      for (int r = 0; r < 8; ++r) mx[r] = fmaxf(mx[r], SWZ_XOR(mx[r], 1));
#pragma unroll
      for (int r = 0; r < 8; ++r) mx[r] = fmaxf(mx[r], SWZ_XOR(mx[r], 2));
#pragma unroll
      for (int r = 0; r < 8; ++r) mx[r] = fmaxf(mx[r], SWZ_XOR(mx[r], 4));
#pragma unroll
      for (int r = 0; r < 8; ++r) mx[r] = fmaxf(mx[r], SWZ_XOR(mx[r], 8));

      // ---- exp2 (raw v_exp_f32), P to LDS, rescale running state ----
      _Float16* pw = sP[wave];
      float a_r[8];
#pragma unroll
      for (int r = 0; r < 8; ++r) {
        const float mn = fmaxf(m_r[r], mx[r]);
        a_r[r] = __builtin_amdgcn_exp2f(m_r[r] - mn);
        m_r[r] = mn;
        const float p0 = __builtin_amdgcn_exp2f(x0[r] - mn);
        const float p1 = __builtin_amdgcn_exp2f(x1[r] - mn);
        pw[(r + 8 * hi) * PSTR + col]      = (_Float16)p0;
        pw[(r + 8 * hi) * PSTR + col + 16] = (_Float16)p1;
      }
#pragma unroll
      for (int n = 0; n < 4; ++n)
#pragma unroll
        for (int r = 0; r < 8; ++r) oacc[n][r] *= a_r[r];

      // ---- P back in A layout (same-wave LDS ops are in order) ----
      V16 pa;
      pa.h[0] = *(const v8h*)(pw + col * PSTR + hi * 16);
      pa.h[1] = *(const v8h*)(pw + col * PSTR + hi * 16 + 8);

      // ---- Preload all 4 V B-fragments, then row-sum + 4 PV WMMAs ----
      V16 bv[4];
#pragma unroll
      for (int n = 0; n < 4; ++n) {
        const _Float16* p = bVp + (n * 16 + col) * VSTR + hi * 16;
        bv[n].h[0] = *(const v8h*)(p);
        bv[n].h[1] = *(const v8h*)(p + 8);
      }
      v8f lz;
#pragma unroll
      for (int i = 0; i < 8; ++i) lz[i] = 0.0f;
      lz = wmma_f16(pa.v, ones, lz);
#pragma unroll
      for (int n = 0; n < 4; ++n) oacc[n] = wmma_f16(pa.v, bv[n].v, oacc[n]);
#pragma unroll
      for (int r = 0; r < 8; ++r) l_r[r] = l_r[r] * a_r[r] + lz[r];
    }

    if (kt + 1 < nkt) stage_tile((kt + 1) & 1);  // write opposite buffer
  }

  // ---- Normalize and store ----
#pragma unroll
  for (int r = 0; r < 8; ++r) {
    const float inv  = 1.0f / l_r[r];
    const int   rowg = qrow0 + r + 8 * hi;
    float* dst = Out + base + (size_t)rowg * D_DIM + col;
#pragma unroll
    for (int n = 0; n < 4; ++n) dst[n * 16] = oacc[n][r] * inv;
  }
}

extern "C" void kernel_launch(void* const* d_in, const int* in_sizes, int n_in,
                              void* d_out, int out_size, void* d_ws, size_t ws_size,
                              hipStream_t stream) {
  (void)in_sizes; (void)n_in; (void)out_size; (void)d_ws; (void)ws_size;
  const float* Q = (const float*)d_in[0];
  const float* K = (const float*)d_in[1];
  const float* V = (const float*)d_in[2];
  // d_in[3] is the causal mask; tril by construction, applied analytically.
  float* O = (float*)d_out;
  dim3 grid(S_LEN / QTILE, 2 * 16);  // (q-blocks, B*H)
  flash_attn_causal_f16wmma<<<grid, 128, 0, stream>>>(Q, K, V, O);
}
